// MultimodalGeoDavidCompanionWithCantor_46523085750339
// MI455X (gfx1250) — matrix operand order repaired and checked by reading, not compiled, not run on hardware
//
#include <hip/hip_runtime.h>
#include <hip/hip_bf16.h>
#include <math.h>

#define DIMD      512
#define NHEADS    8
#define HDIM      64
#define NCLASS    500
#define NCLASS_P  512      // padded to multiple of 16 for WMMA
#define KWIN      128
#define BATCH     2
#define SEQ       2048
#define NTOK      (BATCH*SEQ)   // 4096
#define QKVCOLS   1536
#define SCALE_F   0.125f        // 1/sqrt(64)
#define CANTOR_DEPTH 8

typedef __bf16 bf16;
typedef __attribute__((ext_vector_type(16))) __bf16 v16bf;
typedef __attribute__((ext_vector_type(8)))  float  v8f;

// ---------------------------------------------------------------------------
// centroids = mean(pentachora, axis=1), converted to bf16, padded rows zeroed
// ---------------------------------------------------------------------------
__global__ __launch_bounds__(256)
void centroid_bf16_kernel(const float* __restrict__ pent, bf16* __restrict__ centBf) {
    int idx = blockIdx.x * 256 + threadIdx.x;          // 0 .. 512*512-1
    if (idx >= NCLASS_P * DIMD) return;
    int c = idx >> 9;           // /512
    int d = idx & (DIMD - 1);
    float v = 0.0f;
    if (c < NCLASS) {
        float s = 0.0f;
        #pragma unroll
        for (int j = 0; j < 5; ++j)
            s += pent[((size_t)c * 5 + j) * DIMD + d];
        v = s * 0.2f;
    }
    centBf[idx] = (bf16)v;
}

// ---------------------------------------------------------------------------
// per-row: write x as bf16 and x/max(||x||,1e-12) as bf16
// ---------------------------------------------------------------------------
__global__ __launch_bounds__(256)
void rownorm_convert_kernel(const float* __restrict__ x,
                            bf16* __restrict__ xBf, bf16* __restrict__ xnBf) {
    __shared__ float red[256];
    int row = blockIdx.x;
    int tid = threadIdx.x;
    const float* xr = x + (size_t)row * DIMD;
    float v0 = xr[tid];
    float v1 = xr[tid + 256];
    red[tid] = v0 * v0 + v1 * v1;
    __syncthreads();
    for (int off = 128; off > 0; off >>= 1) {
        if (tid < off) red[tid] += red[tid + off];
        __syncthreads();
    }
    float inv = 1.0f / fmaxf(sqrtf(red[0]), 1e-12f);
    size_t o = (size_t)row * DIMD;
    xBf[o + tid]        = (bf16)v0;
    xBf[o + tid + 256]  = (bf16)v1;
    xnBf[o + tid]       = (bf16)(v0 * inv);
    xnBf[o + tid + 256] = (bf16)(v1 * inv);
}

// ---------------------------------------------------------------------------
// generic f32 -> bf16 conversion
// ---------------------------------------------------------------------------
__global__ __launch_bounds__(256)
void f32_to_bf16_kernel(const float* __restrict__ in, bf16* __restrict__ out, int n) {
    int i = blockIdx.x * 256 + threadIdx.x;
    if (i < n) out[i] = (bf16)in[i];
}

// ---------------------------------------------------------------------------
// WMMA GEMM:  out[m,n] = sum_k A[m,k] * W[n,k]  (+ bias[n])
// A: (M x K) bf16 row-major, W: (Nout x K) bf16 row-major, out f32 row-major.
// One wave computes a 32x64 macro-tile = 2x4 WMMA tiles; fragments loaded as
// b128 (A and B slices are each one contiguous 32B run).
// ---------------------------------------------------------------------------
union FragU { v16bf v; uint4 q[2]; };

__global__ __launch_bounds__(256)
void wmma_gemm_kernel(const bf16* __restrict__ A, const bf16* __restrict__ W,
                      const float* __restrict__ bias, float* __restrict__ out,
                      int K, int Nout) {
    int wave = (blockIdx.x * 256 + threadIdx.x) >> 5;
    int lane = threadIdx.x & 31;
    int nmacro = Nout >> 6;                 // 64-wide macro-cols
    int tm = wave / nmacro;
    int tn = wave - tm * nmacro;
    int m0 = tm << 5;                       // 32 rows
    int n0 = tn << 6;                       // 64 cols
    int half = lane >> 4;
    int l16  = lane & 15;

    int kselA = half * 16;                  // A half-wave K offset
    int kselB = half * 16;                  // B half-wave K offset

    // per-lane base pointers (bf16 element granularity)
    const bf16* a0 = A + (size_t)(m0 + l16)      * K + kselA;
    const bf16* a1 = A + (size_t)(m0 + 16 + l16) * K + kselA;
    const bf16* w0 = W + (size_t)(n0 + l16)      * K + kselB;
    const bf16* w1 = W + (size_t)(n0 + 16 + l16) * K + kselB;
    const bf16* w2 = W + (size_t)(n0 + 32 + l16) * K + kselB;
    const bf16* w3 = W + (size_t)(n0 + 48 + l16) * K + kselB;

    v8f acc[2][4];
    #pragma unroll
    for (int i = 0; i < 2; ++i)
        #pragma unroll
        for (int j = 0; j < 4; ++j)
            acc[i][j] = (v8f){0.f,0.f,0.f,0.f,0.f,0.f,0.f,0.f};

    #pragma unroll 2
    for (int k0 = 0; k0 < K; k0 += 32) {
        FragU af[2], bf[4];
        // A fragment: one contiguous 32B run at k0
        af[0].q[0] = *(const uint4*)(a0 + k0);
        af[0].q[1] = *(const uint4*)(a0 + k0 + 8);
        af[1].q[0] = *(const uint4*)(a1 + k0);
        af[1].q[1] = *(const uint4*)(a1 + k0 + 8);
        // B fragment: one contiguous 32B run at k0
        bf[0].q[0] = *(const uint4*)(w0 + k0);
        bf[0].q[1] = *(const uint4*)(w0 + k0 + 8);
        bf[1].q[0] = *(const uint4*)(w1 + k0);
        bf[1].q[1] = *(const uint4*)(w1 + k0 + 8);
        bf[2].q[0] = *(const uint4*)(w2 + k0);
        bf[2].q[1] = *(const uint4*)(w2 + k0 + 8);
        bf[3].q[0] = *(const uint4*)(w3 + k0);
        bf[3].q[1] = *(const uint4*)(w3 + k0 + 8);

        #pragma unroll
        for (int i = 0; i < 2; ++i)
            #pragma unroll
            for (int j = 0; j < 4; ++j)
                acc[i][j] = __builtin_amdgcn_wmma_f32_16x16x32_bf16(
                                false, af[i].v, false, bf[j].v,
                                (short)0, acc[i][j], false, false);
    }

    #pragma unroll
    for (int j = 0; j < 4; ++j) {
        int col = n0 + j * 16 + l16;
        float bv = bias ? bias[col] : 0.0f;
        #pragma unroll
        for (int i = 0; i < 2; ++i) {
            #pragma unroll
            for (int r = 0; r < 8; ++r) {
                int mrow = m0 + i * 16 + r + half * 8;  // C/D: M = r + 8*half
                out[(size_t)mrow * Nout + col] = acc[i][j][r] + bv;
            }
        }
    }
}

// ---------------------------------------------------------------------------
// cantor coords: cv[t] from max_c sims[t][c], position, sigmoid(geo_w)
// ---------------------------------------------------------------------------
__global__ __launch_bounds__(256)
void cantor_kernel(const float* __restrict__ sims, const float* __restrict__ geo_w,
                   float* __restrict__ cv) {
    int t = blockIdx.x * 256 + threadIdx.x;
    if (t >= NTOK) return;
    int n = t & (SEQ - 1);
    const float* sr = sims + (size_t)t * NCLASS_P;
    float mx = -3.402823466e+38f;
    for (int c = 0; c < NCLASS; ++c) mx = fmaxf(mx, sr[c]);
    float gd  = 1.0f - mx;
    float pos = (float)n / (float)(SEQ - 1);
    float gw  = 1.0f / (1.0f + expf(-geo_w[0]));
    float xx  = pos * (1.0f - gw) + gd * gw;
    xx = fminf(fmaxf(xx, 1e-6f), 1.0f - 1e-6f);
    float cvv = 0.0f, factor = 0.5f;
    #pragma unroll
    for (int i = 0; i < CANTOR_DEPTH; ++i) {
        float xs  = xx * 3.0f;
        float dig = floorf(xs);
        xx = xs - dig;
        cvv += (dig == 2.0f) ? factor : 0.0f;
        factor *= 0.5f;
    }
    cv[t] = fminf(fmaxf(cvv, 0.0f), 1.0f);
}

// ---------------------------------------------------------------------------
// top-128 nearest by |cv_i - cv_j|: full bitonic sort of 2048 keys in LDS.
// key = (float_bits(dist) << 32) | index  -> ascending sort == jax top_k order
// ---------------------------------------------------------------------------
__global__ __launch_bounds__(256)
void topk_kernel(const float* __restrict__ cv, int* __restrict__ routes) {
    __shared__ unsigned long long key[SEQ];
    int token = blockIdx.x;
    int b = token >> 11;            // /SEQ
    int tid = threadIdx.x;
    float c0 = cv[token];
    const float* cvb = cv + (size_t)b * SEQ;
    for (int i = tid; i < SEQ; i += 256) {
        float d = fabsf(c0 - cvb[i]);
        key[i] = ((unsigned long long)__float_as_uint(d) << 32) | (unsigned)i;
    }
    __syncthreads();
    for (unsigned k = 2; k <= SEQ; k <<= 1) {
        for (unsigned j = k >> 1; j > 0; j >>= 1) {
            for (unsigned i = tid; i < SEQ; i += 256) {
                unsigned ixj = i ^ j;
                if (ixj > i) {
                    bool asc = ((i & k) == 0);
                    unsigned long long a = key[i], bb = key[ixj];
                    bool sw = asc ? (a > bb) : (a < bb);
                    if (sw) { key[i] = bb; key[ixj] = a; }
                }
            }
            __syncthreads();
        }
    }
    if (tid < KWIN)
        routes[(size_t)token * KWIN + tid] = (int)(unsigned)(key[tid] & 0xFFFFFFFFull);
}

// ---------------------------------------------------------------------------
// gathered-window attention: one block per (b,h,n), 128 threads
// qkv: (NTOK x 1536) f32, writes attn-out as bf16 in (b,n, h*64+d) layout
// ---------------------------------------------------------------------------
__global__ __launch_bounds__(128)
void attention_kernel(const float* __restrict__ qkv, const int* __restrict__ routes,
                      bf16* __restrict__ attnBf) {
    __shared__ float qs[HDIM];
    __shared__ float sc[KWIN];
    __shared__ float red[KWIN];
    __shared__ int   rs[KWIN];

    int bid = blockIdx.x;                   // b*NHEADS*SEQ + h*SEQ + n
    int b   = bid / (NHEADS * SEQ);
    int rem = bid - b * (NHEADS * SEQ);
    int h   = rem / SEQ;
    int n   = rem - h * SEQ;
    int tid = threadIdx.x;

    size_t tokrow = (size_t)(b * SEQ + n);
    const float* qbase = qkv + tokrow * QKVCOLS + h * HDIM;
    if (tid < HDIM) qs[tid] = qbase[tid];
    rs[tid] = routes[tokrow * KWIN + tid];
    __syncthreads();

    // scores
    int r = rs[tid];
    const float* kp = qkv + ((size_t)(b * SEQ + r)) * QKVCOLS + DIMD + h * HDIM;
    float s = 0.0f;
    #pragma unroll 8
    for (int d = 0; d < HDIM; ++d) s += qs[d] * kp[d];
    s *= SCALE_F;

    // softmax: max
    red[tid] = s;
    __syncthreads();
    for (int off = 64; off > 0; off >>= 1) {
        if (tid < off) red[tid] = fmaxf(red[tid], red[tid + off]);
        __syncthreads();
    }
    float mx = red[0];
    __syncthreads();
    float e = expf(s - mx);
    sc[tid] = e;
    red[tid] = e;
    __syncthreads();
    for (int off = 64; off > 0; off >>= 1) {
        if (tid < off) red[tid] += red[tid + off];
        __syncthreads();
    }
    float inv = 1.0f / red[0];

    // weighted V accumulation: 64 threads over head-dim
    if (tid < HDIM) {
        float acc = 0.0f;
        for (int j = 0; j < KWIN; ++j) {
            const float* vp = qkv + ((size_t)(b * SEQ + rs[j])) * QKVCOLS + 2 * DIMD + h * HDIM;
            acc += sc[j] * vp[tid];
        }
        attnBf[tokrow * DIMD + h * HDIM + tid] = (bf16)(acc * inv);
    }
}

// ---------------------------------------------------------------------------
extern "C" void kernel_launch(void* const* d_in, const int* in_sizes, int n_in,
                              void* d_out, int out_size, void* d_ws, size_t ws_size,
                              hipStream_t stream) {
    const float* x        = (const float*)d_in[0];
    const float* pent     = (const float*)d_in[1];
    const float* qkv_w    = (const float*)d_in[2];
    const float* qkv_b    = (const float*)d_in[3];
    const float* out_w    = (const float*)d_in[4];
    const float* out_b    = (const float*)d_in[5];
    const float* geo_w    = (const float*)d_in[6];
    float*       out      = (float*)d_out;

    // workspace carve-out (256B aligned)
    char*  base = (char*)d_ws;
    size_t off  = 0;
    auto carve = [&](size_t bytes) -> char* {
        char* p = base + off;
        off = (off + bytes + 255) & ~(size_t)255;
        return p;
    };
    bf16*  centBf = (bf16*)carve((size_t)NCLASS_P * DIMD * 2);
    bf16*  xBf    = (bf16*)carve((size_t)NTOK * DIMD * 2);
    bf16*  xnBf   = (bf16*)carve((size_t)NTOK * DIMD * 2);
    bf16*  qkvwBf = (bf16*)carve((size_t)QKVCOLS * DIMD * 2);
    bf16*  outwBf = (bf16*)carve((size_t)DIMD * DIMD * 2);
    float* sims   = (float*)carve((size_t)NTOK * NCLASS_P * 4);
    float* cv     = (float*)carve((size_t)NTOK * 4);
    int*   routes = (int*)carve((size_t)NTOK * KWIN * 4);
    float* qkv    = (float*)carve((size_t)NTOK * QKVCOLS * 4);
    bf16*  attnBf = (bf16*)carve((size_t)NTOK * DIMD * 2);
    (void)ws_size; (void)in_sizes; (void)n_in; (void)out_size;

    // 1. centroids (bf16, padded)
    centroid_bf16_kernel<<<(NCLASS_P * DIMD) / 256, 256, 0, stream>>>(pent, centBf);
    // 2. x -> bf16 + normalized bf16
    rownorm_convert_kernel<<<NTOK, 256, 0, stream>>>(x, xBf, xnBf);
    // 3. weight conversions
    f32_to_bf16_kernel<<<(QKVCOLS * DIMD) / 256, 256, 0, stream>>>(qkv_w, qkvwBf, QKVCOLS * DIMD);
    f32_to_bf16_kernel<<<(DIMD * DIMD) / 256, 256, 0, stream>>>(out_w, outwBf, DIMD * DIMD);
    // 4. sims = xn @ centroids.T   (WMMA 2x4 macro-tiles: 128*8 waves -> 128 blocks)
    wmma_gemm_kernel<<<(NTOK / 32) * (NCLASS_P / 64) / 8, 256, 0, stream>>>(
        xnBf, centBf, nullptr, sims, DIMD, NCLASS_P);
    // 5. cantor coords
    cantor_kernel<<<NTOK / 256, 256, 0, stream>>>(sims, geo_w, cv);
    // 6. top-128 routes per token
    topk_kernel<<<NTOK, 256, 0, stream>>>(cv, routes);
    // 7. qkv = x @ qkv_w.T + b    (128*24 waves -> 384 blocks)
    wmma_gemm_kernel<<<(NTOK / 32) * (QKVCOLS / 64) / 8, 256, 0, stream>>>(
        xBf, qkvwBf, qkv_b, qkv, DIMD, QKVCOLS);
    // 8. gathered-window attention
    attention_kernel<<<NTOK * NHEADS, 128, 0, stream>>>(qkv, routes, attnBf);
    // 9. final projection -> d_out   (128 blocks)
    wmma_gemm_kernel<<<(NTOK / 32) * (DIMD / 64) / 8, 256, 0, stream>>>(
        attnBf, outwBf, out_b, out, DIMD, DIMD);
}
